// SelfAttention_36189394436541
// MI455X (gfx1250) — matrix operand order repaired and checked by reading, not compiled, not run on hardware
//
#include <hip/hip_runtime.h>
#include <hip/hip_bf16.h>
#include <math.h>

// ---------------------------------------------------------------------------
// Problem constants (from the reference)
// ---------------------------------------------------------------------------
#define BATCH     32
#define DIM       4096
#define NKV       8
#define NREP      4
#define HD        128
#define KVDIM     (NKV * HD)        // 1024
#define MAXS      4096
#define KSPLIT    4

typedef __attribute__((ext_vector_type(16))) __bf16 v16bf;
typedef __attribute__((ext_vector_type(8)))  float  v8f;

union BF16Frag {
    v16bf v;
    unsigned short s[16];
};

// fp32 -> bf16, round-half-up (2 VALU ops vs 3 for full RNE; bf16-lossy anyway)
static __device__ __forceinline__ unsigned short f2bf(float f) {
    union { float f; unsigned u; } x;
    x.f = f;
    return (unsigned short)((x.u + 0x8000u) >> 16);
}

// ---------------------------------------------------------------------------
// Split-K GEMM partial: P[ks] = X[:, kspan] @ W[:, kspan]^T
// X: 32 x K row-major, W: N x K row-major, P: KSPLIT x 32 x N partials.
// Block = 128 threads (4 waves), wave owns a 32x16 slab. Grid = (N/64, KSPLIT).
// bf16 WMMA 16x16x32, f32 accumulate.
// ---------------------------------------------------------------------------
__global__ void gemm_xwt_wmma_split(const float* __restrict__ X,
                                    const float* __restrict__ W,
                                    float* __restrict__ P,
                                    int N, int K) {
    const int lane = threadIdx.x & 31;
    const int wave = threadIdx.x >> 5;
    const int n0   = blockIdx.x * 64 + wave * 16;
    const int r    = lane & 15;   // row (A) / col (B/C) within tile
    const int h    = lane >> 4;   // lane-half selects K sub-range

    const int kspan = K / KSPLIT;
    const int kbase = blockIdx.y * kspan;

    const float* xrow0 = X + (size_t)r        * K;  // M-tile 0 (rows 0..15)
    const float* xrow1 = X + (size_t)(r + 16) * K;  // M-tile 1 (rows 16..31)
    const float* wrow  = W + (size_t)(n0 + r) * K;  // B column n0+r == W row

    v8f c0 = {};
    v8f c1 = {};

    for (int k = kbase; k < kbase + kspan; k += 32) {
        __builtin_prefetch(wrow + k + 128, 0, 1);   // global_prefetch_b8
        BF16Frag a0, a1, b;
        // A (16-bit A 16x32 ISA layout): elems 0..7 -> K=k+8h+e,
        //                                elems 8..15 -> K=k+16+8h+e
        const int ka = k + 8 * h;
#pragma unroll
        for (int e = 0; e < 8; ++e) {
            a0.s[e]     = f2bf(xrow0[ka + e]);
            a0.s[e + 8] = f2bf(xrow0[ka + 16 + e]);
            a1.s[e]     = f2bf(xrow1[ka + e]);
            a1.s[e + 8] = f2bf(xrow1[ka + 16 + e]);
        }
        // B: same K layout as A: elems 0..7 -> K=k+8h+e, elems 8..15 -> K=k+16+8h+e
        const int kb = k + 8 * h;
#pragma unroll
        for (int e = 0; e < 8; ++e) {
            b.s[e]     = f2bf(wrow[kb + e]);
            b.s[e + 8] = f2bf(wrow[kb + 16 + e]);
        }

        c0 = __builtin_amdgcn_wmma_f32_16x16x32_bf16(
                false, a0.v, false, b.v, (short)0, c0, false, false);
        c1 = __builtin_amdgcn_wmma_f32_16x16x32_bf16(
                false, a1.v, false, b.v, (short)0, c1, false, false);
    }

    // C/D layout: elem v -> row (v + 8h), col (n0 + r)
    float* Pp = P + (size_t)blockIdx.y * 32 * N;
#pragma unroll
    for (int v = 0; v < 8; ++v) {
        const int m = v + 8 * h;
        Pp[(size_t)m        * N + n0 + r] = c0[v];
        Pp[(size_t)(m + 16) * N + n0 + r] = c1[v];
    }
}

// Deterministic split-K reduction: Y[i] = sum_ks P[ks][i]
__global__ void reduce_ks(const float* __restrict__ P,
                          float* __restrict__ Y,
                          int total) {
    const int i = blockIdx.x * 256 + threadIdx.x;
    if (i >= total) return;
    float s = P[i];
#pragma unroll
    for (int ks = 1; ks < KSPLIT; ++ks)
        s += P[(size_t)ks * total + i];
    Y[i] = s;
}

// ---------------------------------------------------------------------------
// RoPE over 32 x (8 heads x 128) in place (reference ropes both xk and xv).
// ---------------------------------------------------------------------------
__global__ void rope_kernel(float* __restrict__ kv,
                            const float* __restrict__ cosv,
                            const float* __restrict__ sinv) {
    const int idx = blockIdx.x * blockDim.x + threadIdx.x; // 32*8*64 = 16384
    if (idx >= BATCH * NKV * (HD / 2)) return;
    const int i  = idx & 63;
    const int bh = idx >> 6;
    const float c = cosv[i];
    const float s = sinv[i];
    float* p = kv + (size_t)bh * HD + 2 * i;
    const float re = p[0];
    const float im = p[1];
    p[0] = re * c - im * s;
    p[1] = re * s + im * c;
}

// ---------------------------------------------------------------------------
// Flash-decode attention with ASYNCcnt double-buffered K/V staging in LDS.
// One block (8 waves, 256 threads) per (batch, kv-head). Each wave owns an
// independent online-softmax segment; tiles of 2 keys are streamed with
// global_load_async_to_lds_b128 (4 issues/tile), s_wait_asynccnt <= 4 pipelines
// tile t+1's copy under tile t's compute. LDS merge across waves at the end.
// ---------------------------------------------------------------------------
__global__ void attn_kernel(const float* __restrict__ q,     // 32 x 4096
                            const float* __restrict__ knew,  // 32 x 1024
                            const float* __restrict__ vnew,  // 32 x 1024
                            const float* __restrict__ kc,    // 32x4096x8x128
                            const float* __restrict__ vc,    // 32x4096x8x128
                            const int*   __restrict__ sp,
                            float* __restrict__ ctx) {       // 32 x 4096
    const int b    = blockIdx.x >> 3;
    const int hk   = blockIdx.x & 7;
    const int lane = threadIdx.x & 31;
    const int wave = threadIdx.x >> 5;

    const int start_pos = *sp;
    const int T = start_pos + 1;

    // Per-wave double-buffered staging: [wave][buf][row][128]
    // rows: 0=K(key0) 1=V(key0) 2=K(key1) 3=V(key1).  32 KB total.
    __shared__ float stage[8][2][4][HD];
    __shared__ float sm[8];
    __shared__ float sl[8];
    __shared__ float sacc[8][NREP * HD];   // 16 KB

    // q fragments: 4 query heads share this kv head; lane owns dims 4L..4L+3
    float4 ql[4];
#pragma unroll
    for (int g = 0; g < 4; ++g)
        ql[g] = *(const float4*)(q + (size_t)b * DIM + (hk * NREP + g) * HD
                                   + lane * 4);

    const float scale = 0.08838834764831845f; // 1/sqrt(128)

    const float* knew_p = knew + (size_t)b * KVDIM + hk * HD;
    const float* vnew_p = vnew + (size_t)b * KVDIM + hk * HD;
    const float* kbase  = kc + ((size_t)b * MAXS * NKV + hk) * HD;
    const float* vbase  = vc + ((size_t)b * MAXS * NKV + hk) * HD;

    float  m = -3.0e38f;
    float  l = 0.0f;
    float4 acc[4];
#pragma unroll
    for (int g = 0; g < 4; ++g) acc[g] = make_float4(0.f, 0.f, 0.f, 0.f);

    const int per = (T + 7) >> 3;
    const int j0  = wave * per;
    const int j1  = (j0 + per < T) ? (j0 + per) : T;
    const int nt  = (j1 - j0 + 1) >> 1;          // 2-key tiles for this wave

    const unsigned lds_lane = (unsigned)lane * 16u;

    // Issue async copy of tile starting at key jt into buffer `buf`.
    auto issue_tile = [&](int jt, int buf) {
        const unsigned base =
            (unsigned)(unsigned long long)(&stage[wave][buf][0][0]);
#pragma unroll
        for (int rr = 0; rr < 4; ++rr) {
            int key = jt + (rr >> 1);
            if (key >= j1) key = j1 - 1;          // pad slot (never read)
            const float* src;
            if (rr & 1)
                src = (key == start_pos) ? vnew_p
                                         : vbase + (size_t)key * KVDIM;
            else
                src = (key == start_pos) ? knew_p
                                         : kbase + (size_t)key * KVDIM;
            const unsigned lds_off =
                base + (unsigned)rr * (HD * 4u) + lds_lane;
            const unsigned long long gaddr =
                (unsigned long long)(src + lane * 4);
            asm volatile("global_load_async_to_lds_b128 %0, %1, off"
                         :: "v"(lds_off), "v"(gaddr) : "memory");
        }
    };

    auto process_key = [&](const float* krow, const float* vrow) {
        const float4 k4 = *(const float4*)(krow + lane * 4);
        float s0 = ql[0].x*k4.x + ql[0].y*k4.y + ql[0].z*k4.z + ql[0].w*k4.w;
        float s1 = ql[1].x*k4.x + ql[1].y*k4.y + ql[1].z*k4.z + ql[1].w*k4.w;
        float s2 = ql[2].x*k4.x + ql[2].y*k4.y + ql[2].z*k4.z + ql[2].w*k4.w;
        float s3 = ql[3].x*k4.x + ql[3].y*k4.y + ql[3].z*k4.z + ql[3].w*k4.w;
#pragma unroll
        for (int off = 16; off >= 1; off >>= 1) {
            s0 += __shfl_xor(s0, off, 32);
            s1 += __shfl_xor(s1, off, 32);
            s2 += __shfl_xor(s2, off, 32);
            s3 += __shfl_xor(s3, off, 32);
        }
        s0 *= scale; s1 *= scale; s2 *= scale; s3 *= scale;

        const float mx = fmaxf(fmaxf(s0, s1), fmaxf(s2, s3));
        const float mn = fmaxf(m, mx);
        const float corr = __expf(m - mn);
        const float p0 = __expf(s0 - mn);
        const float p1 = __expf(s1 - mn);
        const float p2 = __expf(s2 - mn);
        const float p3 = __expf(s3 - mn);
        l = l * corr + (p0 + p1 + p2 + p3);

        const float4 v4 = *(const float4*)(vrow + lane * 4);
        acc[0].x = acc[0].x*corr + p0*v4.x; acc[0].y = acc[0].y*corr + p0*v4.y;
        acc[0].z = acc[0].z*corr + p0*v4.z; acc[0].w = acc[0].w*corr + p0*v4.w;
        acc[1].x = acc[1].x*corr + p1*v4.x; acc[1].y = acc[1].y*corr + p1*v4.y;
        acc[1].z = acc[1].z*corr + p1*v4.z; acc[1].w = acc[1].w*corr + p1*v4.w;
        acc[2].x = acc[2].x*corr + p2*v4.x; acc[2].y = acc[2].y*corr + p2*v4.y;
        acc[2].z = acc[2].z*corr + p2*v4.z; acc[2].w = acc[2].w*corr + p2*v4.w;
        acc[3].x = acc[3].x*corr + p3*v4.x; acc[3].y = acc[3].y*corr + p3*v4.y;
        acc[3].z = acc[3].z*corr + p3*v4.z; acc[3].w = acc[3].w*corr + p3*v4.w;
        m = mn;
    };

    if (nt > 0) {
        issue_tile(j0, 0);
        for (int t = 0; t < nt; ++t) {
            const int jt  = j0 + 2 * t;
            const int buf = t & 1;
            if (t + 1 < nt) {
                issue_tile(jt + 2, (t + 1) & 1);
                // next tile's 4 copies outstanding -> current tile complete
                asm volatile("s_wait_asynccnt 0x4" ::: "memory");
            } else {
                asm volatile("s_wait_asynccnt 0x0" ::: "memory");
            }
            process_key(&stage[wave][buf][0][0], &stage[wave][buf][1][0]);
            if (jt + 1 < j1)
                process_key(&stage[wave][buf][2][0], &stage[wave][buf][3][0]);
        }
    }

    // Cross-wave combine through LDS.
    if (lane == 0) { sm[wave] = m; sl[wave] = l; }
#pragma unroll
    for (int g = 0; g < 4; ++g)
        *(float4*)&sacc[wave][g * HD + lane * 4] = acc[g];
    __syncthreads();

    float gm = -3.0e38f;
#pragma unroll
    for (int w = 0; w < 8; ++w) gm = fmaxf(gm, sm[w]);
    float gl = 0.0f;
#pragma unroll
    for (int w = 0; w < 8; ++w) gl += sl[w] * __expf(sm[w] - gm);
    const float inv = 1.0f / gl;

    for (int o = threadIdx.x; o < NREP * HD; o += 256) {
        float sum = 0.0f;
#pragma unroll
        for (int w = 0; w < 8; ++w)
            sum += sacc[w][o] * __expf(sm[w] - gm);
        const int g = o >> 7;
        const int d = o & 127;
        ctx[(size_t)b * DIM + (hk * NREP + g) * HD + d] = sum * inv;
    }
}

// ---------------------------------------------------------------------------
// Host launcher
// ---------------------------------------------------------------------------
extern "C" void kernel_launch(void* const* d_in, const int* in_sizes, int n_in,
                              void* d_out, int out_size, void* d_ws, size_t ws_size,
                              hipStream_t stream) {
    const float* x    = (const float*)d_in[0];   // 32 x 4096
    const int*   sp   = (const int*)  d_in[1];   // start_pos
    const float* cosv = (const float*)d_in[2];   // 64
    const float* sinv = (const float*)d_in[3];   // 64
    const float* wq   = (const float*)d_in[4];   // 4096 x 4096
    const float* wk   = (const float*)d_in[5];   // 1024 x 4096
    const float* wv   = (const float*)d_in[6];   // 1024 x 4096
    const float* wo   = (const float*)d_in[7];   // 4096 x 4096
    const float* kc   = (const float*)d_in[8];   // 32 x 4096 x 8 x 128
    const float* vc   = (const float*)d_in[9];
    float*       out  = (float*)d_out;           // 32 x 4096

    float* ws   = (float*)d_ws;
    float* q_ws = ws;                            // 32*4096 = 131072
    float* k_ws = q_ws + BATCH * DIM;            // 32*1024 =  32768
    float* v_ws = k_ws + BATCH * KVDIM;          //            32768
    float* ctx  = v_ws + BATCH * KVDIM;          //           131072
    float* part = ctx  + BATCH * DIM;            // KSPLIT*32*4096 = 524288

    // --- QKV projections: split-K WMMA partials + deterministic reduce ---
    hipLaunchKernelGGL(gemm_xwt_wmma_split, dim3(DIM / 64, KSPLIT), dim3(128),
                       0, stream, x, wq, part, DIM, DIM);
    hipLaunchKernelGGL(reduce_ks, dim3((BATCH * DIM + 255) / 256), dim3(256),
                       0, stream, part, q_ws, BATCH * DIM);

    hipLaunchKernelGGL(gemm_xwt_wmma_split, dim3(KVDIM / 64, KSPLIT), dim3(128),
                       0, stream, x, wk, part, KVDIM, DIM);
    hipLaunchKernelGGL(reduce_ks, dim3((BATCH * KVDIM + 255) / 256), dim3(256),
                       0, stream, part, k_ws, BATCH * KVDIM);

    hipLaunchKernelGGL(gemm_xwt_wmma_split, dim3(KVDIM / 64, KSPLIT), dim3(128),
                       0, stream, x, wv, part, KVDIM, DIM);
    hipLaunchKernelGGL(reduce_ks, dim3((BATCH * KVDIM + 255) / 256), dim3(256),
                       0, stream, part, v_ws, BATCH * KVDIM);

    // --- RoPE on new k and v (reference ropes both) ---
    hipLaunchKernelGGL(rope_kernel, dim3(64), dim3(256), 0, stream,
                       k_ws, cosv, sinv);
    hipLaunchKernelGGL(rope_kernel, dim3(64), dim3(256), 0, stream,
                       v_ws, cosv, sinv);

    // --- Flash-decode attention (async LDS staging) ---
    hipLaunchKernelGGL(attn_kernel, dim3(BATCH * NKV), dim3(256), 0, stream,
                       q_ws, k_ws, v_ws, kc, vc, sp, ctx);

    // --- Output projection ---
    hipLaunchKernelGGL(gemm_xwt_wmma_split, dim3(DIM / 64, KSPLIT), dim3(128),
                       0, stream, ctx, wo, part, DIM, DIM);
    hipLaunchKernelGGL(reduce_ks, dim3((BATCH * DIM + 255) / 256), dim3(256),
                       0, stream, part, out, BATCH * DIM);
}
